// PCEncoder_12601434046673
// MI455X (gfx1250) — matrix operand order repaired and checked by reading, not compiled, not run on hardware
//
#include <hip/hip_runtime.h>
#include <hip/hip_bf16.h>

typedef __attribute__((ext_vector_type(16))) _Float16 v16h;
typedef __attribute__((ext_vector_type(8)))  _Float16 v8h;
typedef __attribute__((ext_vector_type(8)))  float    v8f;

namespace {
constexpr int BATCH = 8;
constexpr int NPTS  = 20480;
constexpr int MA    = 256;
constexpr int MB    = 256;
constexpr int KI    = 3;
constexpr int KAB   = 32;
constexpr int NP    = BATCH * NPTS;          // 163840 point columns
constexpr int COLS_KNN = BATCH * MB * KAB;   // 65536 knn columns
constexpr int COLS_NB  = BATCH * MB;         // 2048 node_b columns

// d_out layout (floats, reference tuple order, flattened)
constexpr size_t O_PCC    = 0;                                   // pc_centers  (B,3,N)
constexpr size_t O_CMEAN  = O_PCC    + (size_t)BATCH * 3 * NPTS; // cluster_mean(B,3,Ma)
constexpr size_t O_MKI    = O_CMEAN  + (size_t)BATCH * 3 * MA;   // min_k_idx   (B,N,3) as float
constexpr size_t O_FIRST  = O_MKI    + (size_t)BATCH * NPTS * KI;// first_pn_out(B,64,N)
constexpr size_t O_SECOND = O_FIRST  + (size_t)BATCH * 64 * NPTS;// second_pn_out(B,128,N)
constexpr size_t O_NAF    = O_SECOND + (size_t)BATCH * 128 * NPTS;// node_a_features(B,128,Ma)
constexpr size_t O_NBF    = O_NAF    + (size_t)BATCH * 128 * MA; // node_b_features(B,256,Mb)
constexpr size_t O_GLOB   = O_NBF    + (size_t)BATCH * 256 * MB; // global_feature(B,1024,1)
} // namespace

// ---------------------------------------------------------------------------
__global__ __launch_bounds__(256) void k_zero(float* __restrict__ p, size_t n) {
  size_t i = (size_t)blockIdx.x * blockDim.x + threadIdx.x;
  size_t stride = (size_t)gridDim.x * blockDim.x;
  for (; i < n; i += stride) p[i] = 0.f;
}

// weight pre-conversion: f32 [O][Creal] -> f16 [O][Cpad], zero-padded
__global__ __launch_bounds__(256) void k_cvt_w(
    const float* __restrict__ W, _Float16* __restrict__ out,
    int O, int Creal, int Cpad) {
  int i = blockIdx.x * 256 + threadIdx.x;
  if (i >= O * Cpad) return;
  int o = i / Cpad, c = i - o * Cpad;
  out[i] = (_Float16)((c < Creal) ? W[(size_t)o * Creal + c] : 0.f);
}

// ---------------------------------------------------------------------------
// 3-NN of each point against node_a; segment count/sum accumulation
__global__ __launch_bounds__(256) void k_nn3(
    const float* __restrict__ pc, const float* __restrict__ node_a,
    float* __restrict__ mki_out, int* __restrict__ minidx,
    float* __restrict__ counts, float* __restrict__ sums) {
  __shared__ float na[3 * MA];
  const int blocksPerB = NPTS / 256;               // 80
  const int b = blockIdx.x / blocksPerB;
  const int n = (blockIdx.x % blocksPerB) * 256 + threadIdx.x;
  for (int i = threadIdx.x; i < 3 * MA; i += 256)
    na[i] = node_a[(size_t)b * 3 * MA + i];
  __syncthreads();
  const float px = pc[((size_t)b * 3 + 0) * NPTS + n];
  const float py = pc[((size_t)b * 3 + 1) * NPTS + n];
  const float pz = pc[((size_t)b * 3 + 2) * NPTS + n];
  float d0 = 1e30f, d1 = 1e30f, d2 = 1e30f;
  int i0 = 0, i1 = 0, i2 = 0;
  for (int m = 0; m < MA; ++m) {
    float dx = px - na[0 * MA + m];
    float dy = py - na[1 * MA + m];
    float dz = pz - na[2 * MA + m];
    float dd = dx * dx + dy * dy + dz * dz;
    if (dd < d0)      { d2 = d1; i2 = i1; d1 = d0; i1 = i0; d0 = dd; i0 = m; }
    else if (dd < d1) { d2 = d1; i2 = i1; d1 = dd; i1 = m; }
    else if (dd < d2) { d2 = dd; i2 = m; }
  }
  const size_t gp = (size_t)b * NPTS + n;
  mki_out[gp * 3 + 0] = (float)i0;
  mki_out[gp * 3 + 1] = (float)i1;
  mki_out[gp * 3 + 2] = (float)i2;
  minidx[gp] = i0;
  atomicAdd(&counts[(size_t)b * MA + i0], 1.f);
  atomicAdd(&sums[((size_t)b * 3 + 0) * MA + i0], px);
  atomicAdd(&sums[((size_t)b * 3 + 1) * MA + i0], py);
  atomicAdd(&sums[((size_t)b * 3 + 2) * MA + i0], pz);
}

// cluster_mean = sums / (counts + 1e-5)
__global__ __launch_bounds__(256) void k_mean(
    const float* __restrict__ counts, const float* __restrict__ sums,
    float* __restrict__ cmean) {
  int i = blockIdx.x * 256 + threadIdx.x;          // over B*MA
  if (i >= BATCH * MA) return;
  int b = i / MA, m = i - b * MA;
  float c = counts[i] + 1e-5f;
  for (int d = 0; d < 3; ++d)
    cmean[((size_t)b * 3 + d) * MA + m] = sums[((size_t)b * 3 + d) * MA + m] / c;
}

// build pc_aug (7 ch -> pad 32), column-major [col][32]; emit pc_centers
__global__ __launch_bounds__(256) void k_pack_pcaug(
    const float* __restrict__ pc, const float* __restrict__ inten,
    const float* __restrict__ sn, const float* __restrict__ cmean,
    const int* __restrict__ minidx, float* __restrict__ pcc_out,
    _Float16* __restrict__ X) {
  int gp = blockIdx.x * 256 + threadIdx.x;
  if (gp >= NP) return;
  int b = gp / NPTS, n = gp - b * NPTS;
  int a = minidx[gp];
  _Float16* xc = X + (size_t)gp * 32;
#pragma unroll
  for (int c = 0; c < 3; ++c) {
    float ctr = cmean[((size_t)b * 3 + c) * MA + a];
    float p = pc[((size_t)b * 3 + c) * NPTS + n];
    pcc_out[((size_t)b * 3 + c) * NPTS + n] = ctr;
    xc[c] = (_Float16)(p - ctr);
  }
  xc[3] = (_Float16)inten[(size_t)b * NPTS + n];
#pragma unroll
  for (int c = 0; c < 3; ++c)
    xc[4 + c] = (_Float16)sn[((size_t)b * 3 + c) * NPTS + n];
#pragma unroll
  for (int c = 7; c < 32; ++c) xc[c] = (_Float16)0.f;
}

// gathered first_max into channels 64..127 of second_pn input [col][128]
__global__ __launch_bounds__(256) void k_pack_second(
    const float* __restrict__ first_max, const int* __restrict__ minidx,
    _Float16* __restrict__ X) {
  int gp = blockIdx.x * 256 + threadIdx.x;
  if (gp >= NP) return;
  int b = gp / NPTS;
  int a = minidx[gp];
  _Float16* xc = X + (size_t)gp * 128 + 64;
  for (int o = 0; o < 64; ++o)
    xc[o] = (_Float16)first_max[((size_t)b * 64 + o) * MA + a];
}

// top-32 NN of node_b against cluster_mean
__global__ __launch_bounds__(256) void k_knn32(
    const float* __restrict__ cmean, const float* __restrict__ node_b,
    int* __restrict__ knnidx) {
  __shared__ float cm[3 * MA];
  int b = blockIdx.x;
  int mb = threadIdx.x;
  for (int i = threadIdx.x; i < 3 * MA; i += 256)
    cm[i] = cmean[(size_t)b * 3 * MA + i];
  __syncthreads();
  float x = node_b[((size_t)b * 3 + 0) * MB + mb];
  float y = node_b[((size_t)b * 3 + 1) * MB + mb];
  float z = node_b[((size_t)b * 3 + 2) * MB + mb];
  unsigned used[MA / 32] = {0, 0, 0, 0, 0, 0, 0, 0};
  for (int kk = 0; kk < KAB; ++kk) {
    float best = 1e30f; int bi = 0;
    for (int m = 0; m < MA; ++m) {
      if ((used[m >> 5] >> (m & 31)) & 1u) continue;
      float dx = cm[0 * MA + m] - x;
      float dy = cm[1 * MA + m] - y;
      float dz = cm[2 * MA + m] - z;
      float d = dx * dx + dy * dy + dz * dz;
      if (d < best) { best = d; bi = m; }
    }
    used[bi >> 5] |= 1u << (bi & 31);
    knnidx[((size_t)b * MB + mb) * KAB + kk] = bi;
  }
}

// knn_before input (131 ch -> pad 160), column-major [col][160]
__global__ __launch_bounds__(256) void k_pack_knn(
    const float* __restrict__ cmean, const float* __restrict__ node_b,
    const float* __restrict__ naf, const int* __restrict__ knnidx,
    _Float16* __restrict__ X) {
  int col = blockIdx.x * 256 + threadIdx.x;        // over 65536
  if (col >= COLS_KNN) return;
  int b = col >> 13;                               // / (MB*KAB)
  int rem = col & (MB * KAB - 1);
  int mb = rem >> 5;
  int a = knnidx[col];
  _Float16* xc = X + (size_t)col * 160;
#pragma unroll
  for (int c = 0; c < 3; ++c)
    xc[c] = (_Float16)(cmean[((size_t)b * 3 + c) * MA + a] -
                       node_b[((size_t)b * 3 + c) * MB + mb]);
  for (int c = 0; c < 128; ++c)
    xc[3 + c] = (_Float16)naf[((size_t)b * 128 + c) * MA + a];
#pragma unroll
  for (int c = 131; c < 160; ++c) xc[c] = (_Float16)0.f;
}

// pooled knn features -> knn_after input [col][256]
__global__ __launch_bounds__(256) void k_pack_after(
    const float* __restrict__ xmax, _Float16* __restrict__ X) {
  int col = blockIdx.x * 256 + threadIdx.x;        // over 2048
  if (col >= COLS_NB) return;
  int b = col >> 8, mb = col & 255;
  _Float16* xc = X + (size_t)col * 256;
  for (int c = 0; c < 256; ++c)
    xc[c] = (_Float16)xmax[((size_t)b * 256 + c) * MB + mb];
}

// concat(node_b, node_b_features) (259 -> pad 288) -> final_pn input [col][288]
__global__ __launch_bounds__(256) void k_pack_final(
    const float* __restrict__ node_b, const float* __restrict__ nbf,
    _Float16* __restrict__ X) {
  int col = blockIdx.x * 256 + threadIdx.x;        // over 2048
  if (col >= COLS_NB) return;
  int b = col >> 8, mb = col & 255;
  _Float16* xc = X + (size_t)col * 288;
#pragma unroll
  for (int c = 0; c < 3; ++c)
    xc[c] = (_Float16)node_b[((size_t)b * 3 + c) * MB + mb];
  for (int c = 0; c < 256; ++c)
    xc[3 + c] = (_Float16)nbf[((size_t)b * 256 + c) * MB + mb];
#pragma unroll
  for (int c = 259; c < 288; ++c) xc[c] = (_Float16)0.f;
}

// ---------------------------------------------------------------------------
// WMMA GEMM: Y[o,col] = relu(sum_c Wf[o][c]*X[col][c] + bias[o])
// Wf is pre-converted f16, zero-padded [O][Cpad].
// Weight tiles are double-buffered through LDS with
// global_load_async_to_lds_b128: the DMA for chunk i+1 overlaps the A-fragment
// loads and 4 WMMAs of chunk i (one barrier per chunk).
// A = activation tile (16 points x 32 K): per-lane two contiguous v8h global
//     loads (contiguous K order: K = h*16 + s, same as B).
// B = Wf^T tile from LDS: lane nl reads Wl[o=nl][h*16..h*16+15] (2x b128).
// D: lane holds out-channel nl, points colTile + 8h + v (contiguous in v).
// amode: 0 none; 1 segment max via segIdx[col]; 2 k-group max; 3 global max.
__global__ __launch_bounds__(256) void k_gemm(
    const _Float16* __restrict__ Wf, const float* __restrict__ bias,
    int O, int Cpad,
    const _Float16* __restrict__ Xin,
    _Float16* __restrict__ Yh, int YhStride,
    float* __restrict__ Yf, int colsPerBatch,
    float* __restrict__ Amax, int amode,
    const int* __restrict__ segIdx, int segCount) {
  __shared__ __align__(64) _Float16 Wl[2][64 * 32];
  const int t = threadIdx.x;
  const int lane = t & 31;
  const int wv = t >> 5;
  const int h = lane >> 4;
  const int nl = lane & 15;
  const int oBase = blockIdx.y * 64;
  const int colTile = blockIdx.x * 128 + wv * 16;  // this wave's 16 point-cols

  v8f acc[4];
  v8f z = {};
  acc[0] = z; acc[1] = z; acc[2] = z; acc[3] = z;

  const _Float16* xcol = Xin + (size_t)(colTile + nl) * Cpad;
  // per-thread 16B slice of the 64x32 weight tile
  const int i0 = t * 8;
  const int wr = i0 >> 5, wc = i0 & 31;
  const _Float16* wsrc = Wf + (size_t)(oBase + wr) * Cpad + wc;
  const unsigned ldsOff0 = (unsigned)(size_t)&Wl[0][i0]; // LDS addr = addr[31:0]
  const unsigned ldsOff1 = (unsigned)(size_t)&Wl[1][i0];
  const int nch = Cpad >> 5;

  // prologue: DMA chunk 0 into buffer 0
  asm volatile("global_load_async_to_lds_b128 %0, %1, off"
               :: "v"(ldsOff0), "v"(wsrc) : "memory");

  for (int ci = 0; ci < nch; ++ci) {
    asm volatile("s_wait_asynccnt 0x0" ::: "memory"); // chunk ci landed
    __syncthreads();                                  // ..for every wave
    if (ci + 1 < nch) {                               // DMA next chunk into
      const _Float16* src = wsrc + ((ci + 1) << 5);   //   the other buffer
      unsigned dst = ((ci + 1) & 1) ? ldsOff1 : ldsOff0;
      asm volatile("global_load_async_to_lds_b128 %0, %1, off"
                   :: "v"(dst), "v"(src) : "memory");
    }
    const int kb = ci << 5;
    const _Float16* wbuf = Wl[ci & 1];

    // A fragment: two aligned contiguous 16B loads from global
    v8h alo = *(const v8h*)(xcol + kb + h * 16);
    v8h ahi = *(const v8h*)(xcol + kb + h * 16 + 8);
    v16h afr = __builtin_shufflevector(alo, ahi, 0, 1, 2, 3, 4, 5, 6, 7,
                                       8, 9, 10, 11, 12, 13, 14, 15);
#pragma unroll
    for (int ot = 0; ot < 4; ++ot) {
      // B fragment: 32 contiguous, 32B-aligned bytes of LDS weight row
      v16h bfr = *(const v16h*)&wbuf[(ot * 16 + nl) * 32 + h * 16];
      acc[ot] = __builtin_amdgcn_wmma_f32_16x16x32_f16(
          false, afr, false, bfr, (short)0, acc[ot], false, false);
    }
  }

  // epilogue
  const int b = colTile / colsPerBatch;            // whole tile in one batch
  const int col0 = colTile + 8 * h;                // points for v = 0..7
#pragma unroll
  for (int ot = 0; ot < 4; ++ot) {
    const int o = oBase + ot * 16 + nl;
    const float bv = bias[o];
    float yv[8];
#pragma unroll
    for (int v = 0; v < 8; ++v) {
      float y = acc[ot][v] + bv;
      yv[v] = y > 0.f ? y : 0.f;
    }
    if (Yh) {
#pragma unroll
      for (int v = 0; v < 8; ++v)
        Yh[(size_t)(col0 + v) * YhStride + o] = (_Float16)yv[v];
    }
    if (Yf) {
      float* dst = &Yf[((size_t)b * O + o) * colsPerBatch +
                       (col0 - b * colsPerBatch)];
#pragma unroll
      for (int v = 0; v < 8; ++v) dst[v] = yv[v];   // contiguous -> wide stores
    }
    if (Amax) {
      if (amode == 1) {
#pragma unroll
        for (int v = 0; v < 8; ++v) {
          int seg = segIdx[col0 + v];
          atomicMax((unsigned*)&Amax[((size_t)b * O + o) * segCount + seg],
                    __float_as_uint(yv[v]));
        }
      } else if (amode == 2) {
#pragma unroll
        for (int v = 0; v < 8; ++v) {
          int mb = ((col0 + v) >> 5) & (MB - 1);
          atomicMax((unsigned*)&Amax[((size_t)b * O + o) * segCount + mb],
                    __float_as_uint(yv[v]));
        }
      } else {                                      // amode 3: global max
        float mx = yv[0];
#pragma unroll
        for (int v = 1; v < 8; ++v) mx = yv[v] > mx ? yv[v] : mx;
        atomicMax((unsigned*)&Amax[(size_t)b * O + o], __float_as_uint(mx));
      }
    }
  }
}

// ---------------------------------------------------------------------------
extern "C" void kernel_launch(void* const* d_in, const int* in_sizes, int n_in,
                              void* d_out, int out_size, void* d_ws, size_t ws_size,
                              hipStream_t stream) {
  (void)in_sizes; (void)n_in; (void)out_size; (void)ws_size;
  const float* pc     = (const float*)d_in[0];
  const float* inten  = (const float*)d_in[1];
  const float* sn     = (const float*)d_in[2];
  const float* node_a = (const float*)d_in[3];
  const float* node_b = (const float*)d_in[4];
  // params in setup_inputs() dict insertion order, (w,b) per layer
  const float* Wp[11] = {
    (const float*)d_in[5],  (const float*)d_in[7],  (const float*)d_in[9],
    (const float*)d_in[11], (const float*)d_in[13],
    (const float*)d_in[15], (const float*)d_in[17],
    (const float*)d_in[19], (const float*)d_in[21],
    (const float*)d_in[23], (const float*)d_in[25] };
  const float* Bp[11] = {
    (const float*)d_in[6],  (const float*)d_in[8],  (const float*)d_in[10],
    (const float*)d_in[12], (const float*)d_in[14],
    (const float*)d_in[16], (const float*)d_in[18],
    (const float*)d_in[20], (const float*)d_in[22],
    (const float*)d_in[24], (const float*)d_in[26] };
  const int Osz[11]   = {64, 64, 64, 128, 128, 256, 256, 512, 256, 512, 1024};
  const int Creal[11] = { 7, 64, 64, 128, 128, 131, 256, 256, 512, 259, 512};
  const int Cpad_[11] = {32, 64, 64, 128, 128, 160, 256, 256, 512, 288, 512};

  float* out = (float*)d_out;
  float* o_pcc   = out + O_PCC;
  float* o_cmean = out + O_CMEAN;
  float* o_mki   = out + O_MKI;
  float* o_first = out + O_FIRST;
  float* o_second= out + O_SECOND;
  float* o_naf   = out + O_NAF;
  float* o_nbf   = out + O_NBF;
  float* o_glob  = out + O_GLOB;

  // workspace carve-out (~90 MB total)
  char* w = (char*)d_ws;
  float* counts    = (float*)w;                 w += (size_t)BATCH * MA * 4;
  float* sums      = (float*)w;                 w += (size_t)BATCH * 3 * MA * 4;
  float* first_max = (float*)w;                 w += (size_t)BATCH * 64 * MA * 4;
  float* xmax      = (float*)w;                 w += (size_t)BATCH * 256 * MB * 4;
  int*   minidx    = (int*)w;                   w += (size_t)NP * 4;
  int*   knnidx    = (int*)w;                   w += (size_t)COLS_KNN * 4;
  w = (char*)(((uintptr_t)w + 255) & ~(uintptr_t)255);
  _Float16* wf[11];
  for (int i = 0; i < 11; ++i) {
    wf[i] = (_Float16*)w;
    w += (size_t)Osz[i] * Cpad_[i] * 2;
  }
  w = (char*)(((uintptr_t)w + 255) & ~(uintptr_t)255);
  _Float16* h16a   = (_Float16*)w;              w += (size_t)128 * NP * 2;
  _Float16* h16b   = (_Float16*)w;              /* 128*NP halves */

  // weight pre-conversion (f32 -> zero-padded f16 [O][Cpad])
  for (int i = 0; i < 11; ++i) {
    int nel = Osz[i] * Cpad_[i];
    k_cvt_w<<<(nel + 255) / 256, 256, 0, stream>>>(Wp[i], wf[i],
                                                   Osz[i], Creal[i], Cpad_[i]);
  }

  const size_t nzero_ws = (size_t)BATCH * MA + (size_t)BATCH * 3 * MA +
                          (size_t)BATCH * 64 * MA + (size_t)BATCH * 256 * MB;
  k_zero<<<256, 256, 0, stream>>>(counts, nzero_ws);             // counts..xmax contiguous
  k_zero<<<64, 256, 0, stream>>>(o_naf, (size_t)BATCH * 128 * MA);
  k_zero<<<8, 256, 0, stream>>>(o_glob, (size_t)BATCH * 1024);

  // stage 1: 3-NN, cluster means
  k_nn3<<<NP / 256, 256, 0, stream>>>(pc, node_a, o_mki, minidx, counts, sums);
  k_mean<<<(BATCH * MA + 255) / 256, 256, 0, stream>>>(counts, sums, o_cmean);
  k_pack_pcaug<<<NP / 256, 256, 0, stream>>>(pc, inten, sn, o_cmean, minidx,
                                             o_pcc, h16a);

  // stage 2: first_pn (7->64->64->64) over 163840 cols
  dim3 gP(NP / 128, 1);
  k_gemm<<<gP, 256, 0, stream>>>(wf[0], Bp[0], 64, 32, h16a,
                                 h16b, 64, nullptr, NPTS, nullptr, 0, nullptr, MA);
  k_gemm<<<gP, 256, 0, stream>>>(wf[1], Bp[1], 64, 64, h16b,
                                 h16a, 64, nullptr, NPTS, nullptr, 0, nullptr, MA);
  k_gemm<<<gP, 256, 0, stream>>>(wf[2], Bp[2], 64, 64, h16a,
                                 h16b, 128, o_first, NPTS, first_max, 1, minidx, MA);

  // stage 3: second_pn (128->128->128); input = [first_pn_out ; gathered first_max]
  k_pack_second<<<NP / 256, 256, 0, stream>>>(first_max, minidx, h16b);
  dim3 gP2(NP / 128, 2);
  k_gemm<<<gP2, 256, 0, stream>>>(wf[3], Bp[3], 128, 128, h16b,
                                  h16a, 128, nullptr, NPTS, nullptr, 0, nullptr, MA);
  k_gemm<<<gP2, 256, 0, stream>>>(wf[4], Bp[4], 128, 128, h16a,
                                  nullptr, 0, o_second, NPTS, o_naf, 1, minidx, MA);

  // stage 4: 32-NN node_b -> cluster_mean, knn_before MLP + k-max pooling
  k_knn32<<<BATCH, 256, 0, stream>>>(o_cmean, node_b, knnidx);
  k_pack_knn<<<COLS_KNN / 256, 256, 0, stream>>>(o_cmean, node_b, o_naf, knnidx, h16a);
  dim3 gK(COLS_KNN / 128, 4);
  k_gemm<<<gK, 256, 0, stream>>>(wf[5], Bp[5], 256, 160, h16a,
                                 h16b, 256, nullptr, MB * KAB, nullptr, 0, nullptr, MB);
  k_gemm<<<gK, 256, 0, stream>>>(wf[6], Bp[6], 256, 256, h16b,
                                 nullptr, 0, nullptr, MB * KAB, xmax, 2, nullptr, MB);

  // stage 5: knn_after (256->512->256) over 2048 cols
  k_pack_after<<<COLS_NB / 256, 256, 0, stream>>>(xmax, h16a);
  dim3 gA0(COLS_NB / 128, 8), gA1(COLS_NB / 128, 4);
  k_gemm<<<gA0, 256, 0, stream>>>(wf[7], Bp[7], 512, 256, h16a,
                                  h16b, 512, nullptr, MB, nullptr, 0, nullptr, MB);
  k_gemm<<<gA1, 256, 0, stream>>>(wf[8], Bp[8], 256, 512, h16b,
                                  nullptr, 0, o_nbf, MB, nullptr, 0, nullptr, MB);

  // stage 6: final_pn (259->512->1024) + global max
  k_pack_final<<<COLS_NB / 256, 256, 0, stream>>>(node_b, o_nbf, h16a);
  dim3 gF0(COLS_NB / 128, 8), gF1(COLS_NB / 128, 16);
  k_gemm<<<gF0, 256, 0, stream>>>(wf[9], Bp[9], 512, 288, h16a,
                                  h16b, 512, nullptr, MB, nullptr, 0, nullptr, MB);
  k_gemm<<<gF1, 256, 0, stream>>>(wf[10], Bp[10], 1024, 512, h16b,
                                  nullptr, 0, nullptr, MB, o_glob, 3, nullptr, MB);
}
